// FlashAttention_71751723647806
// MI455X (gfx1250) — matrix operand order repaired and checked by reading, not compiled, not run on hardware
//
#include <hip/hip_runtime.h>
#include <hip/hip_bf16.h>

typedef _Float16 f16;
typedef _Float16 v16h __attribute__((ext_vector_type(16)));
typedef _Float16 v8h  __attribute__((ext_vector_type(8)));
typedef float    v8f  __attribute__((ext_vector_type(8)));

#define D_MODEL 1024
#define SEQ     2048
#define HEADS   16
#define DK      64
#define BATCH   2
#define ROWS    (BATCH*SEQ)   // 4096

__device__ __forceinline__ v16h cat8(v8h lo, v8h hi) {
    return __builtin_shufflevector(lo, hi, 0,1,2,3,4,5,6,7,8,9,10,11,12,13,14,15);
}

__device__ __forceinline__ v8f wmma_f16(v16h a, v16h b, v8f c) {
    // D = A(16x32 f16) * B(32x16 f16) + C(16x16 f32)
    return __builtin_amdgcn_wmma_f32_16x16x32_f16(
        /*neg_a=*/false, a, /*neg_b=*/false, b,
        /*c_mod=*/(short)0, c, /*reuse_a=*/false, /*reuse_b=*/false);
}

__device__ __forceinline__ float rmax16(float x) {
    #pragma unroll
    for (int off = 1; off < 16; off <<= 1) x = fmaxf(x, __shfl_xor(x, off, 32));
    return x;
}
__device__ __forceinline__ float rsum16(float x) {
    #pragma unroll
    for (int off = 1; off < 16; off <<= 1) x += __shfl_xor(x, off, 32);
    return x;
}

// ---------------------------------------------------------------------------
// fp32 -> fp16 conversion (plain)
// ---------------------------------------------------------------------------
__global__ void cvt_f32_to_f16(const float* __restrict__ src, f16* __restrict__ dst, int n) {
    int i = blockIdx.x * blockDim.x + threadIdx.x;
    if (i < n) dst[i] = (f16)src[i];
}

// fp32 [K][N] -> fp16 transposed [N][K]  (coalesced writes)
__global__ void cvt_f32_to_f16_T(const float* __restrict__ src, f16* __restrict__ dst,
                                 int K, int N) {
    int i = blockIdx.x * blockDim.x + threadIdx.x;
    if (i < K * N) {
        int n = i / K, k = i % K;
        dst[i] = (f16)src[(size_t)k * N + n];
    }
}

// ---------------------------------------------------------------------------
// WMMA GEMM, no LDS: one wave computes a 32x64 tile of C = alpha*(A*Bt^T + bias)
// A: [M x K] f16 row-major; Bt: [N x K] f16 row-major (pre-transposed weights)
// layout 1: f16 head-major [(b*16+h)][s][64]
// layout 3: f16 head-major transposed [(b*16+h)][d][s]   (for V)
// layout 2: f32 out [M][N]
// ---------------------------------------------------------------------------
__global__ __launch_bounds__(32)
void gemm32x64(const f16* __restrict__ A, const f16* __restrict__ Bt,
               const float* __restrict__ bias,
               f16* __restrict__ outH, float* __restrict__ outF,
               int M, int N, int K, float alpha, int layout)
{
    const int lane = threadIdx.x;
    const int g = lane >> 4;          // lane group (0/1)
    const int r = lane & 15;
    const int tilesN = N / 64;
    const int m0 = (blockIdx.x / tilesN) * 32;
    const int n0 = (blockIdx.x % tilesN) * 64;

    v8f acc[2][4];
    #pragma unroll
    for (int h = 0; h < 2; ++h)
        #pragma unroll
        for (int nt = 0; nt < 4; ++nt) acc[h][nt] = v8f{};

    for (int k0 = 0; k0 < K; k0 += 32) {
        // Two A fragments (rows m0..m0+15 and m0+16..m0+31)
        const f16* a0 = A + (size_t)(m0 + r) * K + k0;
        const f16* a1 = A + (size_t)(m0 + 16 + r) * K + k0;
        v16h af0 = cat8(*(const v8h*)(a0 + 16 * g), *(const v8h*)(a0 + 16 * g + 8));
        v16h af1 = cat8(*(const v8h*)(a1 + 16 * g), *(const v8h*)(a1 + 16 * g + 8));

        #pragma unroll
        for (int nt = 0; nt < 4; ++nt) {
            // B fragment straight from transposed weights: contiguous 32 bytes
            const f16* bp = Bt + (size_t)(n0 + nt * 16 + r) * K + k0 + 16 * g;
            v16h bf = cat8(*(const v8h*)bp, *(const v8h*)(bp + 8));
            acc[0][nt] = wmma_f16(af0, bf, acc[0][nt]);
            acc[1][nt] = wmma_f16(af1, bf, acc[1][nt]);
        }
    }

    // epilogue: C layout row = v + 8g (+16 for second half), col = nt*16 + r
    #pragma unroll
    for (int h = 0; h < 2; ++h) {
        #pragma unroll
        for (int v = 0; v < 8; ++v) {
            const int row = m0 + 16 * h + v + 8 * g;
            #pragma unroll
            for (int nt = 0; nt < 4; ++nt) {
                const int col = n0 + nt * 16 + r;
                float val = (acc[h][nt][v] + bias[col]) * alpha;
                if (layout == 2) {
                    outF[(size_t)row * N + col] = val;
                } else {
                    const int b = row >> 11, s = row & 2047;
                    const int hd = col >> 6, d = col & 63;
                    if (layout == 1)
                        outH[(((size_t)(b * HEADS + hd) * SEQ) + s) * DK + d] = (f16)val;
                    else  // layout 3: V transposed per head
                        outH[(((size_t)(b * HEADS + hd) * DK) + d) * SEQ + s] = (f16)val;
                }
            }
        }
    }
}

// ---------------------------------------------------------------------------
// Flash attention: one wave per (b*H+h, 16-query tile). Q pre-scaled by 1/8.
// K read as [bh][s][64] (contiguous K^T fragments), V read as [bh][d][s]
// (contiguous P@V fragments). Only LDS use: 1KB P-tile relayout.
// ---------------------------------------------------------------------------
__global__ __launch_bounds__(32)
void flash_attn(const f16* __restrict__ Qh, const f16* __restrict__ Kh,
                const f16* __restrict__ VhT, f16* __restrict__ attn)
{
    __shared__ __align__(16) f16 Pt[16 * 32];   // probs [row][key]

    const int lane = threadIdx.x;
    const int g = lane >> 4, r = lane & 15;
    const int q0 = (blockIdx.x & 127) * 16;
    const int bh = blockIdx.x >> 7;             // 0..31
    const f16* Qb = Qh  + ((size_t)bh * SEQ + q0) * DK;
    const f16* Kb = Kh  + (size_t)bh * SEQ * DK;
    const f16* Vb = VhT + (size_t)bh * DK * SEQ;

    // Q fragments for the two K-steps of Dk=64
    v16h qf[2];
    #pragma unroll
    for (int ks = 0; ks < 2; ++ks) {
        const f16* qrow = Qb + (size_t)r * DK + ks * 32;
        qf[ks] = cat8(*(const v8h*)(qrow + 16 * g),
                      *(const v8h*)(qrow + 16 * g + 8));
    }

    float m[8], l[8];
    v8f acc[4] = {v8f{}, v8f{}, v8f{}, v8f{}};
    #pragma unroll
    for (int v = 0; v < 8; ++v) { m[v] = -INFINITY; l[v] = 0.0f; }

    for (int kt = 0; kt < SEQ; kt += 32) {
        // scores: two 16x16 tiles (keys kt+0..15, kt+16..31), B straight from global
        v8f sc[2];
        #pragma unroll
        for (int nt = 0; nt < 2; ++nt) {
            v8f s = v8f{};
            #pragma unroll
            for (int ks = 0; ks < 2; ++ks) {
                // B[k][n] = Kmat[kt+nt*16+n][ks*32+k]  -> contiguous per lane
                const f16* kr = Kb + (size_t)(kt + nt * 16 + r) * DK + ks * 32 + 16 * g;
                v16h bf = cat8(*(const v8h*)kr, *(const v8h*)(kr + 8));
                s = wmma_f16(qf[ks], bf, s);
            }
            sc[nt] = s;
        }

        // online softmax: row = v + 8g, col = lane r within group
        float escale[8];
        #pragma unroll
        for (int v = 0; v < 8; ++v) {
            float cm = rmax16(fmaxf(sc[0][v], sc[1][v]));
            float mn = fmaxf(m[v], cm);
            escale[v] = __expf(m[v] - mn);
            m[v] = mn;
            float p0 = __expf(sc[0][v] - mn);
            float p1 = __expf(sc[1][v] - mn);
            sc[0][v] = p0; sc[1][v] = p1;
            l[v] = l[v] * escale[v] + rsum16(p0 + p1);
            #pragma unroll
            for (int dt = 0; dt < 4; ++dt) acc[dt][v] *= escale[v];
        }

        // C-layout probs -> A-layout fragment via 1KB LDS round trip
        __syncthreads();   // previous iteration's Pt reads complete
        #pragma unroll
        for (int nt = 0; nt < 2; ++nt)
            #pragma unroll
            for (int v = 0; v < 8; ++v)
                Pt[(v + 8 * g) * 32 + nt * 16 + r] = (f16)sc[nt][v];
        __syncthreads();
        const f16* prow = Pt + r * 32;
        v16h pf = cat8(*(const v8h*)(prow + 16 * g),
                       *(const v8h*)(prow + 16 * g + 8));

        // P (16x32) @ V (32x64): B[k][n] = V[kt+k][dt*16+n] = VhT[dt*16+n][kt+k]
        #pragma unroll
        for (int dt = 0; dt < 4; ++dt) {
            const f16* vp = Vb + (size_t)(dt * 16 + r) * SEQ + kt + 16 * g;
            v16h vf = cat8(*(const v8h*)vp, *(const v8h*)(vp + 8));
            acc[dt] = wmma_f16(pf, vf, acc[dt]);
        }
    }

    // epilogue: normalize and scatter to flat [b*SEQ+s][h*64+d] f16
    const int head = bh & 15, b = bh >> 4;
    #pragma unroll
    for (int v = 0; v < 8; ++v) {
        const float rinv = 1.0f / l[v];
        const int srow = q0 + v + 8 * g;
        const size_t base = ((size_t)(b * SEQ + srow)) * D_MODEL + head * DK;
        #pragma unroll
        for (int dt = 0; dt < 4; ++dt)
            attn[base + dt * 16 + r] = (f16)(acc[dt][v] * rinv);
    }
}

// ---------------------------------------------------------------------------
extern "C" void kernel_launch(void* const* d_in, const int* in_sizes, int n_in,
                              void* d_out, int out_size, void* d_ws, size_t ws_size,
                              hipStream_t stream) {
    const float* x  = (const float*)d_in[0];
    const float* Wq = (const float*)d_in[1];
    const float* bq = (const float*)d_in[2];
    const float* Wk = (const float*)d_in[3];
    const float* bk = (const float*)d_in[4];
    const float* Wv = (const float*)d_in[5];
    const float* bv = (const float*)d_in[6];
    const float* Wo = (const float*)d_in[7];
    const float* bo = (const float*)d_in[8];
    float* out = (float*)d_out;

    char* w = (char*)d_ws;
    const size_t MB = 1024ull * 1024ull;
    f16* xh   = (f16*)(w + 0 * MB);    // 8 MB  [4096][1024]
    f16* WqT  = (f16*)(w + 8 * MB);    // 2 MB  [N][K]
    f16* WkT  = (f16*)(w + 10 * MB);
    f16* WvT  = (f16*)(w + 12 * MB);
    f16* WoT  = (f16*)(w + 14 * MB);
    f16* Qh   = (f16*)(w + 16 * MB);   // 8 MB  [32][2048][64]
    f16* Kh   = (f16*)(w + 24 * MB);   // 8 MB  [32][2048][64]
    f16* VhT  = (f16*)(w + 32 * MB);   // 8 MB  [32][64][2048]
    f16* attn = (f16*)(w + 40 * MB);   // 8 MB  [4096][1024]

    const int nx = ROWS * D_MODEL;     // 4,194,304
    const int nw = D_MODEL * D_MODEL;  // 1,048,576
    cvt_f32_to_f16<<<(nx + 255) / 256, 256, 0, stream>>>(x, xh, nx);
    cvt_f32_to_f16_T<<<(nw + 255) / 256, 256, 0, stream>>>(Wq, WqT, D_MODEL, D_MODEL);
    cvt_f32_to_f16_T<<<(nw + 255) / 256, 256, 0, stream>>>(Wk, WkT, D_MODEL, D_MODEL);
    cvt_f32_to_f16_T<<<(nw + 255) / 256, 256, 0, stream>>>(Wv, WvT, D_MODEL, D_MODEL);
    cvt_f32_to_f16_T<<<(nw + 255) / 256, 256, 0, stream>>>(Wo, WoT, D_MODEL, D_MODEL);

    const int gemmGrid = (ROWS / 32) * (D_MODEL / 64);   // 2048 waves
    // Q scaled by 1/sqrt(Dk) = 0.125 (bias inside the scale)
    gemm32x64<<<gemmGrid, 32, 0, stream>>>(xh, WqT, bq, Qh, nullptr,
                                           ROWS, D_MODEL, D_MODEL, 0.125f, 1);
    gemm32x64<<<gemmGrid, 32, 0, stream>>>(xh, WkT, bk, Kh, nullptr,
                                           ROWS, D_MODEL, D_MODEL, 1.0f, 1);
    gemm32x64<<<gemmGrid, 32, 0, stream>>>(xh, WvT, bv, VhT, nullptr,
                                           ROWS, D_MODEL, D_MODEL, 1.0f, 3);

    flash_attn<<<BATCH * HEADS * (SEQ / 16), 32, 0, stream>>>(Qh, Kh, VhT, attn);

    gemm32x64<<<gemmGrid, 32, 0, stream>>>(attn, WoT, bo, nullptr, out,
                                           ROWS, D_MODEL, D_MODEL, 1.0f, 2);
}
